// MGCC_13116830122245
// MI455X (gfx1250) — matrix operand order repaired and checked by reading, not compiled, not run on hardware
//
#include <hip/hip_runtime.h>
#include <hip/hip_bf16.h>

typedef __attribute__((ext_vector_type(16))) __bf16 v16bf;
typedef __attribute__((ext_vector_type(8)))  float  v8f;

#define BATCH 8
#define HH 96
#define WW 96
#define CC 256              // D
#define NN (HH * WW)        // 9216
#define OO (2 * CC)         // 512
#define LNEPS 1e-5f

// ---------------- block reduction helpers (blockDim.x == 256) ----------------
__device__ __forceinline__ float bred_sum(float v, float* s) {
    int t = threadIdx.x;
    s[t] = v; __syncthreads();
#pragma unroll
    for (int off = 128; off > 0; off >>= 1) {
        if (t < off) s[t] += s[t + off];
        __syncthreads();
    }
    float r = s[0]; __syncthreads();
    return r;
}
__device__ __forceinline__ float bred_max(float v, float* s) {
    int t = threadIdx.x;
    s[t] = v; __syncthreads();
#pragma unroll
    for (int off = 128; off > 0; off >>= 1) {
        if (t < off) s[t] = fmaxf(s[t], s[t + off]);
        __syncthreads();
    }
    float r = s[0]; __syncthreads();
    return r;
}
__device__ __forceinline__ float bred_min(float v, float* s) {
    int t = threadIdx.x;
    s[t] = v; __syncthreads();
#pragma unroll
    for (int off = 128; off > 0; off >>= 1) {
        if (t < off) s[t] = fminf(s[t], s[t + off]);
        __syncthreads();
    }
    float r = s[0]; __syncthreads();
    return r;
}

// A-fragment loader: 16x32 bf16 tile, row-major source with row stride `ld`.
// Lane L: row = L&15; VGPR i holds K pair at 16*(i>>2) + 8*(L>>4) + 2*(i&3).
// Per lane this is two contiguous 16B chunks -> lowers to 2x global_load_b128.
__device__ __forceinline__ v16bf load_afrag(const __bf16* __restrict__ A,
                                            size_t ld, int k, int mrow, int half) {
    v16bf af;
#pragma unroll
    for (int i = 0; i < 8; ++i) {
        int bk = 16 * (i >> 2) + 8 * half + 2 * (i & 3);
        union { unsigned u; __bf16 h[2]; } cv;
        cv.u = *(const unsigned*)(A + (size_t)mrow * ld + k + bk);
        af[2 * i] = cv.h[0]; af[2 * i + 1] = cv.h[1];
    }
    return af;
}

// B-fragment loader (contiguous rows): B(k,n) at Bq[(k+L)*ld + n0 + n].
__device__ __forceinline__ v16bf load_bfrag_rows(const __bf16* __restrict__ Bq,
                                                 size_t ld, int k, int n0, int L) {
    v16bf bf;
#pragma unroll
    for (int j = 0; j < 8; ++j) {
        int bk = 16 * (j >> 2) + 8 * (L >> 4) + 2 * (j & 3);
        bf[2 * j]     = Bq[(size_t)(k + bk) * ld + n0 + (L & 15)];
        bf[2 * j + 1] = Bq[(size_t)(k + bk + 1) * ld + n0 + (L & 15)];
    }
    return bf;
}

// ---------------- K0: Wr f32 -> bf16 ----------------
__global__ void k0_cvt(const float* __restrict__ Wr, __bf16* __restrict__ Wrb, int n) {
    int i = blockIdx.x * 256 + threadIdx.x;
    if (i < n) Wrb[i] = (__bf16)Wr[i];
}

// ---------------- K1: dual LayerNorm + query softmax-over-D ----------------
__global__ void __launch_bounds__(256)
k1_ln(const float* __restrict__ x1, const float* __restrict__ x2,
      const float* __restrict__ g1, const float* __restrict__ b1,
      __bf16* __restrict__ n1t /*[B,D,N] bf16*/,
      float*  __restrict__ n2t /*[B,D,N] f32*/,
      __bf16* __restrict__ qry /*[B,D,N] bf16*/) {
    __shared__ float sred[256];
    int row = blockIdx.x;                 // b*N + n
    int b = row / NN, n = row - b * NN;
    int c = threadIdx.x;
    float gg = g1[c], bb = b1[c];

    float v1 = x1[(size_t)row * CC + c];
    float mu1 = bred_sum(v1, sred) * (1.0f / CC);
    float d1 = v1 - mu1;
    float var1 = bred_sum(d1 * d1, sred) * (1.0f / CC);
    float y1 = d1 * rsqrtf(var1 + LNEPS) * gg + bb;
    n1t[((size_t)b * CC + c) * NN + n] = (__bf16)y1;

    float v2 = x2[(size_t)row * CC + c];
    float mu2 = bred_sum(v2, sred) * (1.0f / CC);
    float d2 = v2 - mu2;
    float var2 = bred_sum(d2 * d2, sred) * (1.0f / CC);
    float y2 = d2 * rsqrtf(var2 + LNEPS) * gg + bb;
    n2t[((size_t)b * CC + c) * NN + n] = y2;

    float m = bred_max(y2, sred);
    float e = __expf(y2 - m);
    float se = bred_sum(e, sred);
    qry[((size_t)b * CC + c) * NN + n] = (__bf16)(e / se);
}

// ---------------- K2: key softmax over N (per (b,d) row), 1/sum folded in ----------------
__global__ void __launch_bounds__(256)
k2_key(const float* __restrict__ n2t, __bf16* __restrict__ kexp) {
    __shared__ float sred[256];
    int row = blockIdx.x;                 // b*D + d
    const float* p = n2t + (size_t)row * NN;
    __bf16* q = kexp + (size_t)row * NN;
    int t = threadIdx.x;
    float m = -1e30f;
    for (int i = t; i < NN; i += 256) m = fmaxf(m, p[i]);
    m = bred_max(m, sred);
    float s = 0.0f;
    for (int i = t; i < NN; i += 256) s += __expf(p[i] - m);
    s = bred_sum(s, sred);
    float inv = 1.0f / s;
    for (int i = t; i < NN; i += 256) q[i] = (__bf16)(__expf(p[i] - m) * inv);
}

// ---------------- K3: context[b] = key_sm (DxN) * value^T (NxD), WMMA bf16 ----------------
// grid (e_tiles=16, d_groups=4, B), one wave per 64(d) x 16(e) strip.
// B gather (strided) is double-buffered so its latency hides under 4 WMMAs;
// all 4 A fragments are loaded before the WMMA burst.
__global__ void __launch_bounds__(32)
k3_ctx(const __bf16* __restrict__ kexp, const __bf16* __restrict__ n1t,
       float* __restrict__ ctx) {
    int b = blockIdx.z, dbase = blockIdx.y * 64, e0 = blockIdx.x * 16;
    int L = threadIdx.x & 31;
    int mrow = L & 15, half = L >> 4;
    const __bf16* Bm = n1t + ((size_t)b * CC + e0) * NN;   // B(k,n) = Bm[n*NN + k]
    const __bf16* A0 = kexp + ((size_t)b * CC + dbase) * NN;
    v8f acc[4] = {};

    // prologue: gather first B fragment (strided 16-bit loads)
    v16bf bf;
#pragma unroll
    for (int j = 0; j < 8; ++j) {
        bf[2 * j]     = Bm[(size_t)(L & 15) * NN + 16 * (j >> 2) + 8 * (L >> 4) + 2 * (j & 3)];
        bf[2 * j + 1] = Bm[(size_t)(L & 15) * NN + 16 * (j >> 2) + 8 * (L >> 4) + 2 * (j & 3) + 1];
    }

    for (int k = 0; k < NN; k += 32) {
        // speculative prefetch of next A chunk (global_prefetch_b8; OOB drops silently)
        __builtin_prefetch(A0 + (size_t)mrow * NN + k + 32, 0, 1);

        // issue next-iteration B gather early so it overlaps this step's WMMAs
        int kn = (k + 32 < NN) ? (k + 32) : k;
        v16bf bfn;
#pragma unroll
        for (int j = 0; j < 8; ++j) {
            bfn[2 * j]     = Bm[(size_t)(L & 15) * NN + kn + 16 * (j >> 2) + 8 * (L >> 4) + 2 * (j & 3)];
            bfn[2 * j + 1] = Bm[(size_t)(L & 15) * NN + kn + 16 * (j >> 2) + 8 * (L >> 4) + 2 * (j & 3) + 1];
        }

        // batch all A fragment loads, then fire the WMMA burst
        v16bf afs[4];
#pragma unroll
        for (int t = 0; t < 4; ++t)
            afs[t] = load_afrag(A0 + (size_t)(16 * t) * NN, NN, k, mrow, half);
#pragma unroll
        for (int t = 0; t < 4; ++t)
            acc[t] = __builtin_amdgcn_wmma_f32_16x16x32_bf16(false, afs[t], false, bf,
                                                             (short)0, acc[t], false, false);
        bf = bfn;
    }
#pragma unroll
    for (int t = 0; t < 4; ++t) {
        float* Cp = ctx + ((size_t)b * CC + dbase + 16 * t) * CC + e0;
#pragma unroll
        for (int r = 0; r < 8; ++r)
            Cp[(size_t)(r + 8 * half) * CC + (L & 15)] = acc[t][r];
    }
}

// ---------------- K4: top-k masked softmaxes, combined with a1..a4 -> M bf16 ----------------
__global__ void __launch_bounds__(256)
k4_mask(const float* __restrict__ ctx,
        const float* __restrict__ a1, const float* __restrict__ a2,
        const float* __restrict__ a3, const float* __restrict__ a4,
        __bf16* __restrict__ Mmat) {
    __shared__ float vals[256];
    __shared__ float sred[256];
    int row = blockIdx.x;                 // b*D + d
    int t = threadIdx.x;
    float v = ctx[(size_t)row * CC + t];
    vals[t] = v; __syncthreads();
    int cnt = 0;                          // # strictly greater in row
#pragma unroll 8
    for (int j = 0; j < 256; ++j) cnt += (vals[j] > v) ? 1 : 0;
    __syncthreads();

    const int kk[4] = { CC / 2, (2 * CC) / 3, (3 * CC) / 4, (4 * CC) / 5 }; // 128,170,192,204
    float thr[4];
#pragma unroll
    for (int i = 0; i < 4; ++i)
        thr[i] = bred_min((cnt <= kk[i] - 1) ? v : 1e30f, sred); // k-th largest

    float mx = bred_max(v, sred);
    float ev = __expf(v - mx);
    float as[4] = { a1[0], a2[0], a3[0], a4[0] };
    float w = 0.0f;
#pragma unroll
    for (int i = 0; i < 4; ++i) {
        float den = bred_sum((v >= thr[i]) ? ev : 0.0f, sred);
        if (v >= thr[i]) w += as[i] * ev / den;
    }
    Mmat[(size_t)row * CC + t] = (__bf16)w;
}

// ---------------- K5: att[b] = M (DxD) * query (DxN), WMMA bf16 ----------------
// grid (n_groups=144, d_groups=4, B), block 128 (4 waves = 4 n-tiles).
// Each wave: 64(d) x 16(n) strip; B loaded once per K-step, A fragments batched.
__global__ void __launch_bounds__(128)
k5_att(const __bf16* __restrict__ Mmat, const __bf16* __restrict__ qry,
       __bf16* __restrict__ att) {
    int wave = threadIdx.x >> 5, L = threadIdx.x & 31;
    int b = blockIdx.z, dbase = blockIdx.y * 64;
    int n0 = (blockIdx.x * 4 + wave) * 16;
    int mrow = L & 15, half = L >> 4;
    const __bf16* A0 = Mmat + ((size_t)b * CC + dbase) * CC; // A[m][k]
    const __bf16* Bq = qry  + (size_t)b * CC * NN;           // B(k,n)=Bq[k*NN + n0+n]
    v8f acc[4] = {};
#pragma unroll
    for (int k = 0; k < CC; k += 32) {
        v16bf bf = load_bfrag_rows(Bq, NN, k, n0, L);
        v16bf afs[4];
#pragma unroll
        for (int t = 0; t < 4; ++t)
            afs[t] = load_afrag(A0 + (size_t)(16 * t) * CC, CC, k, mrow, half);
#pragma unroll
        for (int t = 0; t < 4; ++t)
            acc[t] = __builtin_amdgcn_wmma_f32_16x16x32_bf16(false, afs[t], false, bf,
                                                             (short)0, acc[t], false, false);
    }
#pragma unroll
    for (int t = 0; t < 4; ++t) {
        __bf16* Cp = att + ((size_t)b * CC + dbase + 16 * t) * NN + n0;
#pragma unroll
        for (int r = 0; r < 8; ++r)
            Cp[(size_t)(r + 8 * half) * NN + (L & 15)] = (__bf16)acc[t][r];
    }
}

// ---------------- K6: rep = Wr*att + br, LN over 2C, transposed store ----------------
// grid (n_tiles=576, B), block 256 = 8 waves; each wave: 64(o) x 16(n) strip.
__global__ void __launch_bounds__(256)
k6_out(const __bf16* __restrict__ Wrb, const __bf16* __restrict__ att,
       const float* __restrict__ br, const float* __restrict__ g2,
       const float* __restrict__ b2, float* __restrict__ out) {
    __shared__ float rep[OO * 16];        // rep[o*16 + n], 32 KB
    __shared__ float ps[256], ps2[256];
    __shared__ float mu[16], ri[16];

    int b = blockIdx.y, n0 = blockIdx.x * 16;
    int wave = threadIdx.x >> 5, L = threadIdx.x & 31;
    int mrow = L & 15, half = L >> 4;
    const __bf16* Batt = att + (size_t)b * CC * NN;       // B(k,n)=Batt[k*NN + n0+n]
    const __bf16* A0 = Wrb + (size_t)(wave * 64) * CC;    // this wave's 64 o-rows

    v8f acc[4] = {};
#pragma unroll
    for (int k = 0; k < CC; k += 32) {
        v16bf bf = load_bfrag_rows(Batt, NN, k, n0, L);
        v16bf afs[4];
#pragma unroll
        for (int tt = 0; tt < 4; ++tt)
            afs[tt] = load_afrag(A0 + (size_t)(16 * tt) * CC, CC, k, mrow, half);
#pragma unroll
        for (int tt = 0; tt < 4; ++tt)
            acc[tt] = __builtin_amdgcn_wmma_f32_16x16x32_bf16(false, afs[tt], false, bf,
                                                              (short)0, acc[tt], false, false);
    }
#pragma unroll
    for (int tt = 0; tt < 4; ++tt) {
#pragma unroll
        for (int r = 0; r < 8; ++r) {
            int oo = wave * 64 + tt * 16 + r + 8 * half;
            rep[oo * 16 + (L & 15)] = acc[tt][r] + br[oo];
        }
    }
    __syncthreads();

    int col = threadIdx.x & 15, seg = threadIdx.x >> 4;   // 16 segments x 32 rows
    float s = 0.0f, s2 = 0.0f;
#pragma unroll
    for (int r = 0; r < 32; ++r) {
        float x = rep[(seg * 32 + r) * 16 + col];
        s += x; s2 += x * x;
    }
    ps[threadIdx.x] = s; ps2[threadIdx.x] = s2;
    __syncthreads();
    if (threadIdx.x < 16) {
        float ts = 0.0f, ts2 = 0.0f;
#pragma unroll
        for (int g = 0; g < 16; ++g) { ts += ps[g * 16 + threadIdx.x]; ts2 += ps2[g * 16 + threadIdx.x]; }
        float m = ts * (1.0f / OO);
        float var = ts2 * (1.0f / OO) - m * m;
        mu[threadIdx.x] = m;
        ri[threadIdx.x] = rsqrtf(var + LNEPS);
    }
    __syncthreads();
#pragma unroll
    for (int r = 0; r < 32; ++r) {
        int oo = seg * 32 + r;
        float x = rep[oo * 16 + col];
        float y = (x - mu[col]) * ri[col] * g2[oo] + b2[oo];
        out[((size_t)b * OO + oo) * NN + n0 + col] = y;   // [B,2C,H,W]
    }
}

// ---------------- launch ----------------
extern "C" void kernel_launch(void* const* d_in, const int* in_sizes, int n_in,
                              void* d_out, int out_size, void* d_ws, size_t ws_size,
                              hipStream_t stream) {
    const float* x1 = (const float*)d_in[0];
    const float* x2 = (const float*)d_in[1];
    const float* g1 = (const float*)d_in[2];
    const float* b1 = (const float*)d_in[3];
    const float* Wr = (const float*)d_in[4];
    const float* br = (const float*)d_in[5];
    const float* g2 = (const float*)d_in[6];
    const float* b2 = (const float*)d_in[7];
    const float* a1 = (const float*)d_in[8];
    const float* a2 = (const float*)d_in[9];
    const float* a3 = (const float*)d_in[10];
    const float* a4 = (const float*)d_in[11];
    float* out = (float*)d_out;

    char* ws = (char*)d_ws;
    const size_t SZ_BDN_BF = (size_t)BATCH * CC * NN * sizeof(__bf16); // 37.75 MB
    const size_t SZ_BDN_F  = (size_t)BATCH * CC * NN * sizeof(float);  // 75.5 MB
    const size_t SZ_CTX    = (size_t)BATCH * CC * CC * sizeof(float);  // 2 MB
    const size_t SZ_M      = (size_t)BATCH * CC * CC * sizeof(__bf16); // 1 MB
    const size_t SZ_WRB    = (size_t)OO * CC * sizeof(__bf16);         // 256 KB

    size_t off = 0;
    __bf16* n1t  = (__bf16*)(ws + off); off += SZ_BDN_BF;
    float*  n2t  = (float*) (ws + off); off += SZ_BDN_F;
    __bf16* qry  = (__bf16*)(ws + off); off += SZ_BDN_BF;
    __bf16* kexp = (__bf16*)(ws + off); off += SZ_BDN_BF;
    float*  ctx  = (float*) (ws + off); off += SZ_CTX;
    __bf16* Mmat = (__bf16*)(ws + off); off += SZ_M;
    __bf16* attb = (__bf16*)(ws + off); off += SZ_BDN_BF;
    __bf16* Wrb  = (__bf16*)(ws + off); off += SZ_WRB;

    k0_cvt<<<dim3((OO * CC + 255) / 256), dim3(256), 0, stream>>>(Wr, Wrb, OO * CC);
    k1_ln<<<dim3(BATCH * NN), dim3(256), 0, stream>>>(x1, x2, g1, b1, n1t, n2t, qry);
    k2_key<<<dim3(BATCH * CC), dim3(256), 0, stream>>>(n2t, kexp);
    k3_ctx<<<dim3(CC / 16, CC / 64, BATCH), dim3(32), 0, stream>>>(kexp, n1t, ctx);
    k4_mask<<<dim3(BATCH * CC), dim3(256), 0, stream>>>(ctx, a1, a2, a3, a4, Mmat);
    k5_att<<<dim3(NN / 64, CC / 64, BATCH), dim3(128), 0, stream>>>(Mmat, qry, attb);
    k6_out<<<dim3(NN / 16, BATCH), dim3(256), 0, stream>>>(Wrb, attb, br, g2, b2, out);
}
